// KnowledgeEnhancer_43920335569479
// MI455X (gfx1250) — matrix operand order repaired and checked by reading, not compiled, not run on hardware
//
#include <hip/hip_runtime.h>

// KnowledgeEnhancer forward for MI455X (gfx1250, wave32).
//
// out(B x 64) = SM(B x 384) @ M(384 x 64)
//   SM[b, 3c+l] = softmax_l(sign[c,:] * atoms[b, idx[c,:]])
//   M[3c+l, p]  = sign[c,l] * clamp(w[c]) * (idx[c,l] == p)
//
// Phase 1: one softmax per (row, clause), results written into LDS directly
//          in WMMA A-fragment swizzled order (f32).
// Phase 2: v_wmma_f32_16x16x32_f16 with SM split hi+lo f16 (f32-accurate).

typedef __attribute__((ext_vector_type(16))) _Float16 v16h;
typedef __attribute__((ext_vector_type(8)))  float    v8f;

#define C_CLAUSES 128
#define P_PREDS   64
#define KDIM      (C_CLAUSES * 3)        // 384
#define KSTEPS    (KDIM / 32)            // 12
#define NTILES    (P_PREDS / 16)         // 4
#define ROWS      32                     // rows per block (2 waves x 16)

// ws layout:
//   [0, 49152)      : bfrag  _Float16[KSTEPS][NTILES][32 lanes][16 halves]
//   [49152, 51200)  : ctab   int4[128]  {idx0, idx1, idx2, signbits}
//   [51200, 53248)  : cetab  int4[128]  {fragOff0, fragOff1, fragOff2, 0}
#define BFRAG_BYTES (KSTEPS * NTILES * 32 * 16 * 2)
#define CTAB_OFF    BFRAG_BYTES
#define CETAB_OFF   (BFRAG_BYTES + 2048)

// A-fragment swizzle: for k in [0,384), the f32 element offset within one
// wave's SM staging region (independent of row):
//   kk = k/32, km = k%32, h = (km>>3)&1, v = ((km>>4)<<2) + ((km&7)>>1),
//   p = km&1  ->  off = kk*512 + h*256 + 2v + p   (row adds rl*16, wave adds 6144)
__device__ __forceinline__ int frag_off(int k) {
    int kk = k >> 5;
    int km = k & 31;
    int h  = (km >> 3) & 1;
    int v  = ((km >> 4) << 2) + ((km & 7) >> 1);
    int p  = km & 1;
    return kk * 512 + h * 256 + 2 * v + p;
}

// ---------------------------------------------------------------------------
// Setup: build M in LDS; emit pre-swizzled WMMA B fragments + clause tables.
// ---------------------------------------------------------------------------
__global__ __launch_bounds__(256) void ke_setup(
    const float* __restrict__ cw,
    const long long* __restrict__ lidx,
    const int* __restrict__ sb,
    _Float16* __restrict__ bfrag,
    int4* __restrict__ ctab,
    int4* __restrict__ cetab)
{
    __shared__ _Float16 Mh[KDIM * P_PREDS];   // 48 KB
    const int tid = threadIdx.x;

    for (int i = tid; i < KDIM * P_PREDS; i += 256)
        Mh[i] = (_Float16)0.0f;
    __syncthreads();

    for (int e = tid; e < KDIM; e += 256) {
        int   c   = e / 3;
        int   idx = (int)lidx[e];
        float s   = sb[e] ? 1.0f : -1.0f;
        float wv  = fminf(fmaxf(cw[c], 0.0f), 500.0f);
        Mh[e * P_PREDS + idx] = (_Float16)(s * wv);
    }
    __syncthreads();

    if (tid < C_CLAUSES) {
        int4 q;
        q.x = (int)lidx[tid * 3 + 0];
        q.y = (int)lidx[tid * 3 + 1];
        q.z = (int)lidx[tid * 3 + 2];
        q.w = (sb[tid * 3 + 0] & 1) | ((sb[tid * 3 + 1] & 1) << 1) |
              ((sb[tid * 3 + 2] & 1) << 2);
        ctab[tid] = q;
        int4 e;
        e.x = frag_off(tid * 3 + 0);
        e.y = frag_off(tid * 3 + 1);
        e.z = frag_off(tid * 3 + 2);
        e.w = 0;
        cetab[tid] = e;
    }

    // B fragments (B is 32x16 KxN): lane = h*16+n16, group h covers
    // K = h*8+[0..7] and 16+h*8+[0..7]; element s holds K = (s>>3)*16+h*8+(s&7).
    for (int t = tid; t < KSTEPS * NTILES * 32; t += 256) {
        int kk   = t / (NTILES * 32);
        int rem  = t % (NTILES * 32);
        int nt   = rem / 32;
        int lane = rem % 32;
        int n    = nt * 16 + (lane & 15);
        int h    = lane >> 4;
        _Float16* dst = bfrag + (size_t)t * 16;
#pragma unroll
        for (int v = 0; v < 8; ++v)
#pragma unroll
            for (int p = 0; p < 2; ++p) {
                int k = kk * 32 + (v >> 2) * 16 + h * 8 + ((2 * v + p) & 7);
                dst[2 * v + p] = Mh[k * P_PREDS + n];
            }
    }
}

// ---------------------------------------------------------------------------
// Main kernel: 64 threads (2 waves), 32 rows per block.
// ---------------------------------------------------------------------------
__global__ __launch_bounds__(64) void ke_main(
    const float* __restrict__ ga,
    const _Float16* __restrict__ bfrag,
    const int4* __restrict__ ctab,
    const int4* __restrict__ cetab,
    float* __restrict__ out,
    int Btotal)
{
    // SM staging, swizzled to A-fragment order: [wave][kk][lane][16] f32
    __shared__ __align__(16) float smA[2 * KSTEPS * 32 * 16];  // 48 KB
    __shared__ float atoms[ROWS * 65];                         // pad-65: 8.3 KB
    __shared__ int4  cl[C_CLAUSES];                            // 2 KB
    __shared__ int4  ce[C_CLAUSES];                            // 2 KB

    const int tid = threadIdx.x;
    const int R0  = blockIdx.x * ROWS;
    if (R0 >= Btotal) return;

    // Stage 32 rows of atoms (coalesced b128 in, padded scalar LDS out).
    const float4* gsrc = (const float4*)(ga + (size_t)R0 * P_PREDS);
#pragma unroll
    for (int j = 0; j < 8; ++j) {
        int    i4 = tid + j * 64;          // 512 float4 total
        float4 d  = gsrc[i4];
        int row = i4 >> 4;                 // 16 float4 per row
        int col = (i4 & 15) * 4;
        float* a = &atoms[row * 65 + col];
        a[0] = d.x; a[1] = d.y; a[2] = d.z; a[3] = d.w;
    }
#pragma unroll
    for (int j = 0; j < 2; ++j) {
        int c = tid + j * 64;
        cl[c] = ctab[c];
        ce[c] = cetab[c];
    }
    __syncthreads();

    // ---- Phase 1: one softmax per (clause, row); lanes share the clause
    // within an iteration, so clause-table reads are wave-uniform and the
    // pad-65 atom gathers are bank-conflict-free (bank = (row+idx)%64).
    for (int iter = 0; iter < 64; ++iter) {
        int t   = iter * 64 + tid;         // 4096 tasks
        int c   = t >> 5;
        int row = t & 31;
        int4 q = cl[c];
        int4 e = ce[c];
        const float* ar = &atoms[row * 65];
        float s0 = (q.w & 1) ? 1.0f : -1.0f;
        float s1 = (q.w & 2) ? 1.0f : -1.0f;
        float s2 = (q.w & 4) ? 1.0f : -1.0f;
        float l0 = s0 * ar[q.x];
        float l1 = s1 * ar[q.y];
        float l2 = s2 * ar[q.z];
        float m  = fmaxf(fmaxf(l0, l1), l2);
        float e0 = __expf(l0 - m);
        float e1 = __expf(l1 - m);
        float e2 = __expf(l2 - m);
        float inv = __fdividef(1.0f, e0 + e1 + e2);
        int base = (row >> 4) * (KSTEPS * 512) + (row & 15) * 16;
        smA[base + e.x] = e0 * inv;
        smA[base + e.y] = e1 * inv;
        smA[base + e.z] = e2 * inv;
    }
    __syncthreads();

    // ---- Phase 2: WMMA. Each wave owns a 16-row tile.
    const int lane = tid & 31;
    const int w    = tid >> 5;
    const int h    = lane >> 4;
    const int rl   = lane & 15;

    v8f acc[NTILES] = {};

    for (int kk = 0; kk < KSTEPS; ++kk) {
        // Lane's 16 A-fragment values: contiguous 64B in LDS.
        const float4* fp = (const float4*)&smA[(w * KSTEPS + kk) * 512 + lane * 16];
        float sv[16];
        *(float4*)&sv[0]  = fp[0];
        *(float4*)&sv[4]  = fp[1];
        *(float4*)&sv[8]  = fp[2];
        *(float4*)&sv[12] = fp[3];
        v16h ahi = {};
        v16h alo = {};
#pragma unroll
        for (int i = 0; i < 16; ++i) {
            _Float16 hi = (_Float16)sv[i];
            ahi[i] = hi;
            alo[i] = (_Float16)(sv[i] - (float)hi);
        }
#pragma unroll
        for (int nt = 0; nt < NTILES; ++nt) {
            v16h b = ((const v16h*)bfrag)[(size_t)(kk * NTILES + nt) * 32 + lane];
            acc[nt] = __builtin_amdgcn_wmma_f32_16x16x32_f16(
                false, ahi, false, b, (short)0, acc[nt], false, false);
            acc[nt] = __builtin_amdgcn_wmma_f32_16x16x32_f16(
                false, alo, false, b, (short)0, acc[nt], false, false);
        }
    }

    // D layout: VGPR r -> M=r (lanes 0-15), M=r+8 (lanes 16-31); N = lane&15.
    const int rowg = R0 + w * 16 + h * 8;
#pragma unroll
    for (int nt = 0; nt < NTILES; ++nt)
#pragma unroll
        for (int r = 0; r < 8; ++r)
            out[(size_t)(rowg + r) * P_PREDS + nt * 16 + rl] = acc[nt][r];
}

// ---------------------------------------------------------------------------
extern "C" void kernel_launch(void* const* d_in, const int* in_sizes, int n_in,
                              void* d_out, int out_size, void* d_ws, size_t ws_size,
                              hipStream_t stream)
{
    const float*     ga   = (const float*)d_in[0];      // ground_atoms [B*64] f32
    const float*     cw   = (const float*)d_in[1];      // clause_weights [128] f32
    const long long* lidx = (const long long*)d_in[2];  // literal_idx [128*3] i64
    const int*       sb   = (const int*)d_in[3];        // sign_bits [128*3] i32

    float* out = (float*)d_out;
    const int Btotal = in_sizes[0] / P_PREDS;

    _Float16* bfrag = (_Float16*)d_ws;
    int4*     ctab  = (int4*)((char*)d_ws + CTAB_OFF);
    int4*     cetab = (int4*)((char*)d_ws + CETAB_OFF);

    ke_setup<<<1, 256, 0, stream>>>(cw, lidx, sb, bfrag, ctab, cetab);

    const int nblocks = (Btotal + ROWS - 1) / ROWS;
    ke_main<<<nblocks, 64, 0, stream>>>(ga, bfrag, ctab, cetab, out, Btotal);
}
